// PointNet_41841571398371
// MI455X (gfx1250) — matrix operand order repaired and checked by reading, not compiled, not run on hardware
//
#include <hip/hip_runtime.h>

// ============================================================================
// PointNet encoder for MI455X (gfx1250, wave32, WMMA).
//
// Pipeline (all kernels on `stream`, deterministic):
//   1) sample_idx      : deterministic hash+probe sampling of valid pixels
//   2) build_frag0     : gather x/uv -> f16 B-fragments (Cin padded 6->32)
//   3) build_wfrags x5 : f32 weights -> f16 A-fragments (WMMA register layout)
//   4) layers 1..4     : gemm_wmma -> y_raw f32 ; bn_stats (deterministic tree
//                        reduce, double accum) ; build_frags (BN+ReLU -> f16 B-frags)
//   5) layer 5         : gemm_wmma_l5 : WMMA + ds_swizzle butterfly reductions
//                        -> per-block partial sum/sum^2 (deterministic) and
//                        per-(b,ch) max/min via ordered-uint atomics (order-free)
//   6) l5_stats, compute_gf (monotonicity of BN+ReLU: gf from raw max/min)
//   7) heads + head BN (per-channel stats over batch, ds_swizzle reduce)
//
// Biases of BN'd convs cancel under BatchNorm and are skipped (exact).
// ============================================================================

typedef __attribute__((ext_vector_type(16))) _Float16 v16h;
typedef __attribute__((ext_vector_type(8)))  float    v8f;

#define NPTS_TOT   65536      // B*P = 32*2048
#define HW         65536      // 256*256
#define EPSN       1e-5f

// ds_swizzle_b32 group-of-32 XOR mode: offset[15]=0, and_mask=0x1F, or=0, xor=m.
// One DS op per cross-lane exchange, fixed pattern -> deterministic.
#define SWZX(v, m) \
  __uint_as_float((unsigned)__builtin_amdgcn_ds_swizzle( \
      (int)__float_as_uint(v), (((m) << 10) | 0x1F)))

// ---- ordered-uint encoding for deterministic float atomic max/min ----------
__device__ __forceinline__ unsigned encOrd(float f) {
  unsigned u = __float_as_uint(f);
  return (u & 0x80000000u) ? ~u : (u | 0x80000000u);
}
__device__ __forceinline__ float decOrd(unsigned e) {
  unsigned u = (e & 0x80000000u) ? (e & 0x7fffffffu) : ~e;
  return __uint_as_float(u);
}

// A-fragment K map (matches B): lanes 0-15 carry K=0..15;
// lanes 16-31 carry K=16..31.
__device__ __forceinline__ int akmap(int e, int hi) {
  return e + 16 * hi;
}
// B-fragment K map (dense column-major 32x16): lane half e -> K = e + 16*hi.
__device__ __forceinline__ int bkmap(int e, int hi) { return e + 16 * hi; }

// ---------------------------------------------------------------------------
// 1) deterministic point sampling
// ---------------------------------------------------------------------------
__device__ __forceinline__ unsigned pcg(unsigned x) {
  x = x * 747796405u + 2891336453u;
  x = ((x >> ((x >> 28u) + 4u)) ^ x) * 277803737u;
  return (x >> 22u) ^ x;
}

__global__ void sample_idx(const float* __restrict__ mask_uv,
                           const float* __restrict__ face_mask,
                           int* __restrict__ idx) {
  int t = blockIdx.x * 256 + threadIdx.x;          // 0..65535 = b*2048+p
  int b = t >> 11;
  unsigned r = pcg((unsigned)t * 2654435761u ^ 42u) & 65535u;
  const float* mb = mask_uv + (size_t)b * HW;
  int found = (int)r;
  for (int i = 0; i < HW; ++i) {
    unsigned pos = (r + (unsigned)i) & 65535u;
    if (mb[pos] > 0.f && face_mask[pos] > 0.f) { found = (int)pos; break; }
  }
  idx[t] = found;
}

// ---------------------------------------------------------------------------
// 2) input B-fragments: pts[n][c], c: 0..3 feats, 4..5 uv, 6..31 zero pad
// ---------------------------------------------------------------------------
__global__ void __launch_bounds__(128)
build_frag0(const float* __restrict__ x, const int* __restrict__ idx,
            _Float16* __restrict__ frag) {
  int fragId = blockIdx.x * 4 + (threadIdx.x >> 5);     // nb (KB==1)
  int lane = threadIdx.x & 31, col = lane & 15, hi = lane >> 4;
  int n = fragId * 16 + col;
  int b = n >> 11;
  int pix = idx[n];
  float xi = (float)(pix >> 8), yi = (float)(pix & 255);
  v16h out;
#pragma unroll
  for (int e = 0; e < 16; ++e) {
    int c = bkmap(e, hi);
    float v = 0.f;
    if (c < 4)      v = x[((size_t)b * 4 + c) * HW + pix];
    else if (c == 4) v = xi * (1.f / 128.f) - 1.f;
    else if (c == 5) v = yi * (1.f / 128.f) - 1.f;
    out[e] = (_Float16)v;
  }
  ((v16h*)frag)[(size_t)fragId * 32 + lane] = out;
}

// ---------------------------------------------------------------------------
// 3) weight A-fragments (zero-padded to 16 x 32 multiples)
// ---------------------------------------------------------------------------
__global__ void build_wfrags(const float* __restrict__ W, _Float16* __restrict__ frag,
                             int Cout, int Cin, int KB) {
  int fragId = blockIdx.x;
  int mt = fragId / KB, kb = fragId % KB;
  int lane = threadIdx.x, row = lane & 15, hi = lane >> 4;
  int m = mt * 16 + row;
  v16h out;
#pragma unroll
  for (int e = 0; e < 16; ++e) {
    int k = kb * 32 + akmap(e, hi);
    float v = (m < Cout && k < Cin) ? W[(size_t)m * Cin + k] : 0.f;
    out[e] = (_Float16)v;
  }
  ((v16h*)frag)[(size_t)fragId * 32 + lane] = out;
}

// ---------------------------------------------------------------------------
// 4a) generic WMMA GEMM: y[Cout][N] = A(W) * B(h).  4 waves/block, one
//     16-pt n-tile per wave, WM m-tiles per wave (B-fragments reused WM x).
// ---------------------------------------------------------------------------
template <int WM, int KB>
__global__ void __launch_bounds__(128)
gemm_wmma(const _Float16* __restrict__ wfrag, const _Float16* __restrict__ hfrag,
          float* __restrict__ y) {
  const int lane = threadIdx.x & 31, wave = threadIdx.x >> 5;
  const int nt = blockIdx.x * 4 + wave;
  const int col = lane & 15, hi = lane >> 4;
  const int mBase = blockIdx.y * WM;

  v16h bfrag[KB];
  const v16h* bp = (const v16h*)hfrag + (size_t)nt * KB * 32 + lane;
#pragma unroll
  for (int kb = 0; kb < KB; ++kb) bfrag[kb] = bp[kb * 32];

#pragma unroll
  for (int wm = 0; wm < WM; ++wm) {
    v8f acc = {};
    const v16h* ap = (const v16h*)wfrag + (size_t)(mBase + wm) * KB * 32 + lane;
#pragma unroll
    for (int kb = 0; kb < KB; ++kb) {
      v16h a = ap[kb * 32];
      acc = __builtin_amdgcn_wmma_f32_16x16x32_f16(false, a, false, bfrag[kb],
                                                   (short)0, acc, false, false);
    }
    const int m0 = (mBase + wm) * 16 + hi * 8;
    float* yp = y + (size_t)m0 * NPTS_TOT + nt * 16 + col;
#pragma unroll
    for (int r = 0; r < 8; ++r) yp[(size_t)r * NPTS_TOT] = acc[r];
  }
}

// ---------------------------------------------------------------------------
// 4b) per-channel BN stats over N=65536 (deterministic tree reduce, dbl accum)
//     -> fused scale/shift: yhat*g+be == y*scale + shift
// ---------------------------------------------------------------------------
__global__ void __launch_bounds__(256)
bn_stats(const float* __restrict__ y, const float* __restrict__ g,
         const float* __restrict__ be, float* __restrict__ scale,
         float* __restrict__ shift) {
  __shared__ double ssum[256], ssq[256];
  int c = blockIdx.x;
  double s = 0.0, q = 0.0;
  const float* yr = y + (size_t)c * NPTS_TOT;
  for (int n = threadIdx.x; n < NPTS_TOT; n += 256) {
    float v = yr[n];
    s += v; q += (double)v * v;
  }
  ssum[threadIdx.x] = s; ssq[threadIdx.x] = q;
  __syncthreads();
  for (int st = 128; st > 0; st >>= 1) {
    if (threadIdx.x < st) {
      ssum[threadIdx.x] += ssum[threadIdx.x + st];
      ssq[threadIdx.x]  += ssq[threadIdx.x + st];
    }
    __syncthreads();
  }
  if (threadIdx.x == 0) {
    double mu = ssum[0] / NPTS_TOT;
    double var = ssq[0] / NPTS_TOT - mu * mu;
    float rstd = (float)(1.0 / sqrt(var + (double)EPSN));
    float sc = g[c] * rstd;
    scale[c] = sc;
    shift[c] = be[c] - (float)mu * sc;
  }
}

// ---------------------------------------------------------------------------
// 4c) BN+ReLU+f16-convert -> next layer's B-fragments (pre-swizzled)
// ---------------------------------------------------------------------------
__global__ void __launch_bounds__(128)
build_frags(const float* __restrict__ y, const float* __restrict__ scale,
            const float* __restrict__ shift, _Float16* __restrict__ frag, int KB) {
  int fragId = blockIdx.x * 4 + (threadIdx.x >> 5);
  int lane = threadIdx.x & 31, col = lane & 15, hi = lane >> 4;
  int nb = fragId / KB, kb = fragId % KB;
  int n = nb * 16 + col;
  v16h out;
#pragma unroll
  for (int e = 0; e < 16; ++e) {
    int k = kb * 32 + bkmap(e, hi);
    float v = y[(size_t)k * NPTS_TOT + n] * scale[k] + shift[k];
    out[e] = (_Float16)(v > 0.f ? v : 0.f);
  }
  ((v16h*)frag)[(size_t)fragId * 32 + lane] = out;
}

// ---------------------------------------------------------------------------
// 5) layer-5 (128->1024) WMMA + fused reductions; y never materialized.
//    grid = (1024 nblocks of 64 pts, 16 mgroups of 64 ch), 4 waves/block.
//    sum/sum^2: fixed-order ds_swizzle butterflies + per-block partials.
//    max/min : ordered-uint atomics (commutative/idempotent -> deterministic)
// ---------------------------------------------------------------------------
__global__ void __launch_bounds__(128)
gemm_wmma_l5(const _Float16* __restrict__ wfrag, const _Float16* __restrict__ hfrag,
             float* __restrict__ psum, float* __restrict__ psumsq,
             unsigned* __restrict__ gmax, unsigned* __restrict__ gmin) {
  __shared__ float lsum[256], lsq[256], lmx[256], lmn[256];
  const int lane = threadIdx.x & 31, wave = threadIdx.x >> 5;
  const int nt = blockIdx.x * 4 + wave;
  const int col = lane & 15, hi = lane >> 4;
  const int mBase = blockIdx.y * 4;          // 4 m-tiles = 64 channels

  v16h bfrag[4];
  const v16h* bp = (const v16h*)hfrag + (size_t)nt * 4 * 32 + lane;
#pragma unroll
  for (int kb = 0; kb < 4; ++kb) bfrag[kb] = bp[kb * 32];

#pragma unroll
  for (int wm = 0; wm < 4; ++wm) {
    v8f acc = {};
    const v16h* ap = (const v16h*)wfrag + (size_t)(mBase + wm) * 4 * 32 + lane;
#pragma unroll
    for (int kb = 0; kb < 4; ++kb) {
      v16h a = ap[kb * 32];
      acc = __builtin_amdgcn_wmma_f32_16x16x32_f16(false, a, false, bfrag[kb],
                                                   (short)0, acc, false, false);
    }
#pragma unroll
    for (int r = 0; r < 8; ++r) {
      float v = acc[r];
      float s = v, q = v * v, mx = v, mn = v;
      // butterfly across the 16 N-columns (xor masks < 16 keep hi-halves apart)
      s += SWZX(s, 1);  q += SWZX(q, 1);
      mx = fmaxf(mx, SWZX(mx, 1));  mn = fminf(mn, SWZX(mn, 1));
      s += SWZX(s, 2);  q += SWZX(q, 2);
      mx = fmaxf(mx, SWZX(mx, 2));  mn = fminf(mn, SWZX(mn, 2));
      s += SWZX(s, 4);  q += SWZX(q, 4);
      mx = fmaxf(mx, SWZX(mx, 4));  mn = fminf(mn, SWZX(mn, 4));
      s += SWZX(s, 8);  q += SWZX(q, 8);
      mx = fmaxf(mx, SWZX(mx, 8));  mn = fminf(mn, SWZX(mn, 8));
      if (col == 0) {
        int chLocal = wm * 16 + r + 8 * hi;    // 0..63
        lsum[wave * 64 + chLocal] = s;
        lsq [wave * 64 + chLocal] = q;
        lmx [wave * 64 + chLocal] = mx;
        lmn [wave * 64 + chLocal] = mn;
      }
    }
  }
  __syncthreads();
  if (threadIdx.x < 64) {
    int t = threadIdx.x;
    float s  = lsum[t] + lsum[64 + t] + lsum[128 + t] + lsum[192 + t];
    float q  = lsq [t] + lsq [64 + t] + lsq [128 + t] + lsq [192 + t];
    float mx = fmaxf(fmaxf(lmx[t], lmx[64 + t]), fmaxf(lmx[128 + t], lmx[192 + t]));
    float mn = fminf(fminf(lmn[t], lmn[64 + t]), fminf(lmn[128 + t], lmn[192 + t]));
    int ch = mBase * 16 + t;
    psum  [(size_t)ch * 1024 + blockIdx.x] = s;
    psumsq[(size_t)ch * 1024 + blockIdx.x] = q;
    int b = (blockIdx.x * 64) >> 11;           // 64 pts always inside one batch
    atomicMax(&gmax[b * 1024 + ch], encOrd(mx));
    atomicMin(&gmin[b * 1024 + ch], encOrd(mn));
  }
}

__global__ void init_minmax(unsigned* __restrict__ gmax, unsigned* __restrict__ gmin) {
  int i = blockIdx.x * 256 + threadIdx.x;      // 32768
  gmax[i] = 0u;
  gmin[i] = 0xFFFFFFFFu;
}

__global__ void __launch_bounds__(256)
l5_stats(const float* __restrict__ psum, const float* __restrict__ psumsq,
         const float* __restrict__ g, const float* __restrict__ be,
         float* __restrict__ scale, float* __restrict__ shift) {
  __shared__ double ssum[256], ssq[256];
  int c = blockIdx.x;                          // 1024
  double s = 0.0, q = 0.0;
  for (int i = threadIdx.x; i < 1024; i += 256) {
    s += (double)psum  [(size_t)c * 1024 + i];
    q += (double)psumsq[(size_t)c * 1024 + i];
  }
  ssum[threadIdx.x] = s; ssq[threadIdx.x] = q;
  __syncthreads();
  for (int st = 128; st > 0; st >>= 1) {
    if (threadIdx.x < st) {
      ssum[threadIdx.x] += ssum[threadIdx.x + st];
      ssq[threadIdx.x]  += ssq[threadIdx.x + st];
    }
    __syncthreads();
  }
  if (threadIdx.x == 0) {
    double mu = ssum[0] / NPTS_TOT;
    double var = ssq[0] / NPTS_TOT - mu * mu;
    float rstd = (float)(1.0 / sqrt(var + (double)EPSN));
    float sc = g[c] * rstd;
    scale[c] = sc;
    shift[c] = be[c] - (float)mu * sc;
  }
}

// gf[b][c] = relu(BN(max_P y))  (BN+ReLU monotone; sign(scale) picks max/min)
__global__ void compute_gf(const unsigned* __restrict__ gmax,
                           const unsigned* __restrict__ gmin,
                           const float* __restrict__ scale,
                           const float* __restrict__ shift,
                           float* __restrict__ gf, float* __restrict__ out_gf) {
  int i = blockIdx.x * 256 + threadIdx.x;      // 32768
  int c = i & 1023;
  float sc = scale[c];
  float yraw = (sc >= 0.f) ? decOrd(gmax[i]) : decOrd(gmin[i]);
  float v = yraw * sc + shift[c];
  v = v > 0.f ? v : 0.f;
  gf[i] = v;
  out_gf[i] = v;
}

// ---------------------------------------------------------------------------
// 7) heads: 240 dot products of length 1024 per batch. Tm keeps bias; BN'd
//    heads drop it (cancels). Then per-channel BN over the 32-batch axis.
// ---------------------------------------------------------------------------
__global__ void __launch_bounds__(256)
heads(const float* __restrict__ gf, const float* __restrict__ Wtm,
      const float* __restrict__ btm, const float* __restrict__ Wsh,
      const float* __restrict__ Wex, float* __restrict__ outTm,
      float* __restrict__ shraw, float* __restrict__ exraw) {
  int b = blockIdx.x, o = threadIdx.x;
  if (o >= 240) return;
  const float* gfb = gf + (size_t)b * 1024;
  const float* Wrow;
  if (o < 12)       Wrow = Wtm + (size_t)o * 1024;
  else if (o < 211) Wrow = Wsh + (size_t)(o - 12) * 1024;
  else              Wrow = Wex + (size_t)(o - 211) * 1024;
  const float4* w4 = (const float4*)Wrow;
  const float4* g4 = (const float4*)gfb;
  float s = 0.f;
#pragma unroll 4
  for (int k = 0; k < 256; ++k) {
    float4 a = w4[k], c = g4[k];
    s += a.x * c.x + a.y * c.y + a.z * c.z + a.w * c.w;
  }
  if (o < 12)       outTm[b * 12 + o] = s + btm[o];
  else if (o < 211) shraw[b * 199 + (o - 12)] = s;
  else              exraw[b * 29 + (o - 211)] = s;
}

__global__ void head_bn(const float* __restrict__ raw, const float* __restrict__ g,
                        const float* __restrict__ be, float* __restrict__ out, int O) {
  int c = blockIdx.x, b = threadIdx.x;         // one wave per channel, lane = batch
  float v = raw[b * O + c];
  float s = v, q = v * v;
  s += SWZX(s, 1);   q += SWZX(q, 1);
  s += SWZX(s, 2);   q += SWZX(q, 2);
  s += SWZX(s, 4);   q += SWZX(q, 4);
  s += SWZX(s, 8);   q += SWZX(q, 8);
  s += SWZX(s, 16);  q += SWZX(q, 16);
  float mu = s * (1.f / 32.f);
  float var = q * (1.f / 32.f) - mu * mu;
  float rstd = rsqrtf(var + EPSN);
  out[b * O + c] = (v - mu) * rstd * g[c] + be[c];
}

// ===========================================================================
extern "C" void kernel_launch(void* const* d_in, const int* in_sizes, int n_in,
                              void* d_out, int out_size, void* d_ws, size_t ws_size,
                              hipStream_t stream) {
  (void)in_sizes; (void)n_in; (void)out_size; (void)ws_size;

  const float* X   = (const float*)d_in[0];
  const float* MUV = (const float*)d_in[1];
  const float* FM  = (const float*)d_in[2];
  // params in setup_inputs insertion order: (W,b,g,be) x5, then head W/b, head g/be
  const float* W[5]  = {(const float*)d_in[3],  (const float*)d_in[7],
                        (const float*)d_in[11], (const float*)d_in[15],
                        (const float*)d_in[19]};
  const float* Gm[5] = {(const float*)d_in[5],  (const float*)d_in[9],
                        (const float*)d_in[13], (const float*)d_in[17],
                        (const float*)d_in[21]};
  const float* Be[5] = {(const float*)d_in[6],  (const float*)d_in[10],
                        (const float*)d_in[14], (const float*)d_in[18],
                        (const float*)d_in[22]};
  const float* Wtm = (const float*)d_in[23];
  const float* btm = (const float*)d_in[24];
  const float* Wsh = (const float*)d_in[25];
  const float* Wex = (const float*)d_in[27];
  const float* gsh = (const float*)d_in[29];
  const float* besh= (const float*)d_in[30];
  const float* gex = (const float*)d_in[31];
  const float* beex= (const float*)d_in[32];

  float* OUT = (float*)d_out;                  // Tm[32,12] | shape[32,199] | exp[32,29] | gf[32,1024]
  char* ws = (char*)d_ws;

  // ---- workspace layout (~80 MB) ----
  int*       idx    = (int*)(ws + 0);
  _Float16*  wf[5];
  for (int l = 0; l < 5; ++l) wf[l] = (_Float16*)(ws + (1u << 20) + (size_t)l * (256u << 10));
  float*     scl[5]; float* shf[5];
  for (int l = 0; l < 5; ++l) {
    scl[l] = (float*)(ws + (3u << 20) + (size_t)l * (64u << 10));
    shf[l] = scl[l] + 1024;
  }
  float*     psum   = (float*)(ws + (4u << 20));            // 4 MB  [1024][1024]
  float*     psumsq = (float*)(ws + (8u << 20));            // 4 MB
  unsigned*  gmax   = (unsigned*)(ws + (12u << 20));        // 128 KB
  unsigned*  gmin   = gmax + 32768;
  float*     gf     = (float*)(gmin + 32768);               // 128 KB
  float*     shraw  = gf + 32768;                           // 32*199
  float*     exraw  = shraw + 32 * 199;                     // 32*29
  _Float16*  fragA  = (_Float16*)(ws + (16u << 20));        // 16 MB
  _Float16*  fragB  = (_Float16*)(ws + (32u << 20));        // 16 MB
  float*     ybuf   = (float*)(ws + (48u << 20));           // 32 MB (<=128 ch)

  const int Cout[5] = {64, 64, 64, 128, 1024};
  const int Cin [5] = {6, 64, 64, 64, 128};
  const int KBw [5] = {1, 2, 2, 2, 4};        // padded Cin / 32

  // 0) init deterministic max/min accumulators (harness does not re-poison)
  init_minmax<<<128, 256, 0, stream>>>(gmax, gmin);

  // 1) sampling + input fragments
  sample_idx <<<256, 256, 0, stream>>>(MUV, FM, idx);
  build_frag0<<<1024, 128, 0, stream>>>(X, idx, fragA);

  // 2) weight fragments (A operands, WMMA register layout, f16)
  for (int l = 0; l < 5; ++l) {
    int nf = (Cout[l] / 16) * KBw[l];
    build_wfrags<<<nf, 32, 0, stream>>>(W[l], wf[l], Cout[l], Cin[l], KBw[l]);
  }

  // 3) layers 1-4: GEMM -> stats -> BN+ReLU fragments
  // L1: 32(pad 6) -> 64
  gemm_wmma<4, 1><<<dim3(1024, 1), 128, 0, stream>>>(wf[0], fragA, ybuf);
  bn_stats<<<64, 256, 0, stream>>>(ybuf, Gm[0], Be[0], scl[0], shf[0]);
  build_frags<<<(4096 * 2) / 4, 128, 0, stream>>>(ybuf, scl[0], shf[0], fragB, 2);
  // L2: 64 -> 64
  gemm_wmma<4, 2><<<dim3(1024, 1), 128, 0, stream>>>(wf[1], fragB, ybuf);
  bn_stats<<<64, 256, 0, stream>>>(ybuf, Gm[1], Be[1], scl[1], shf[1]);
  build_frags<<<(4096 * 2) / 4, 128, 0, stream>>>(ybuf, scl[1], shf[1], fragA, 2);
  // L3: 64 -> 64
  gemm_wmma<4, 2><<<dim3(1024, 1), 128, 0, stream>>>(wf[2], fragA, ybuf);
  bn_stats<<<64, 256, 0, stream>>>(ybuf, Gm[2], Be[2], scl[2], shf[2]);
  build_frags<<<(4096 * 2) / 4, 128, 0, stream>>>(ybuf, scl[2], shf[2], fragB, 2);
  // L4: 64 -> 128
  gemm_wmma<4, 2><<<dim3(1024, 2), 128, 0, stream>>>(wf[3], fragB, ybuf);
  bn_stats<<<128, 256, 0, stream>>>(ybuf, Gm[3], Be[3], scl[3], shf[3]);
  build_frags<<<(4096 * 4) / 4, 128, 0, stream>>>(ybuf, scl[3], shf[3], fragA, 4);

  // 4) layer 5: 128 -> 1024, fused stats + max/min (no 256 MB activation)
  gemm_wmma_l5<<<dim3(1024, 16), 128, 0, stream>>>(wf[4], fragA, psum, psumsq, gmax, gmin);
  l5_stats<<<1024, 256, 0, stream>>>(psum, psumsq, Gm[4], Be[4], scl[4], shf[4]);
  compute_gf<<<128, 256, 0, stream>>>(gmax, gmin, scl[4], shf[4], gf, OUT + 7680);

  // 5) heads
  heads<<<32, 256, 0, stream>>>(gf, Wtm, btm, Wsh, Wex, OUT, shraw, exraw);
  head_bn<<<199, 32, 0, stream>>>(shraw, gsh, besh, OUT + 384, 199);
  head_bn<<<29, 32, 0, stream>>>(exraw, gex, beex, OUT + 384 + 6368, 29);
}
